// LSTM_30855045055200
// MI455X (gfx1250) — matrix operand order repaired and checked by reading, not compiled, not run on hardware
//
#include <hip/hip_runtime.h>
#include <hip/hip_bf16.h>
#include <cstdint>
#include <cstddef>

// ---------------------------------------------------------------------------
// LSTM (B=64, S=512, H=1024) for gfx1250 / MI455X.
//   gates = [h, x_t] @ W_all + b  ->  f,i,chat,o  ->  c,h update (512 steps)
// bf16 WMMA (v_wmma_f32_16x16x32_bf16), f32 accumulation.
// Persistent kernel + device-wide atomic barrier; c-state resident in VGPRs.
// h/x slabs staged into LDS by the Tensor Data Mover (tensor_load_to_lds)
// with hardware padding giving a 1032-element LDS row stride (bank-conflict
// free ds_load_b128 A-fragments).
// ---------------------------------------------------------------------------

typedef __attribute__((ext_vector_type(16))) __bf16       v16bf;
typedef __attribute__((ext_vector_type(8)))  float        v8f;
typedef __attribute__((ext_vector_type(4)))  unsigned int v4u;
typedef __attribute__((ext_vector_type(8)))  int          v8i;
typedef __attribute__((ext_vector_type(4)))  int          v4i;

#define HID   1024
#define BATCH 64
#define SEQN  512

// workspace layout (bytes)
#define X_OFF   0ull          // bf16 [S][B][H]      = 64 MB
#define WT_OFF  67108864ull   // bf16 [4*H][2H] W^T  = 16 MB
#define H0_OFF  83886080ull   // bf16 [B][H] ping
#define H1_OFF  84017152ull   // bf16 [B][H] pong
#define HF_OFF  84148224ull   // f32  [B][H] final h
#define BAR_OFF 84410368ull   // unsigned barrier counter

#define LDS_ROW   1032        // padded row stride (elements)
#define SLAB_ELTS (16 * LDS_ROW)

__device__ __forceinline__ float sigf(float x) {
  return __frcp_rn(1.0f + __expf(-x));
}
__device__ __forceinline__ float tanh_fast(float x) {
  float e = __expf(-2.0f * x);
  return (1.0f - e) * __frcp_rn(1.0f + e);
}

// -------------------------------------------------------------------- embed
__global__ __launch_bounds__(256) void embed_cast(const int* __restrict__ xi,
                                                  const float* __restrict__ emb,
                                                  __bf16* __restrict__ xb) {
  size_t fid = (size_t)blockIdx.x * 256 + threadIdx.x;   // float4 units
  int h4 = (int)(fid & 255);
  int bs = (int)(fid >> 8);
  int b  = bs >> 9;
  int s  = bs & 511;
  int idx = xi[b * SEQN + s];
  float4 v = ((const float4*)(emb + (size_t)idx * HID))[h4];
  union { __bf16 h[4]; uint2 u; } pk;
  pk.h[0] = (__bf16)v.x; pk.h[1] = (__bf16)v.y;
  pk.h[2] = (__bf16)v.z; pk.h[3] = (__bf16)v.w;
  *(uint2*)(xb + ((size_t)s * BATCH + b) * HID + (size_t)h4 * 4) = pk.u;
}

// ------------------------------------------------------------- weight pack
// wt[(g*1024 + n)*2048 + k] = (bf16) W_g[k*1024 + n]
__global__ __launch_bounds__(256) void pack_weights(const float* __restrict__ Wf,
                                                    const float* __restrict__ Wi,
                                                    const float* __restrict__ Wc,
                                                    const float* __restrict__ Wo,
                                                    __bf16* __restrict__ wt) {
  int bid = blockIdx.x;
  int nt = bid & 15;
  int kt = (bid >> 4) & 31;
  int g  = bid >> 9;
  const float* W = (g == 0) ? Wf : (g == 1) ? Wi : (g == 2) ? Wc : Wo;
  __shared__ __bf16 tile[64 * 66];
  int c  = threadIdx.x & 63;
  int rg = threadIdx.x >> 6;
  int k0 = kt * 64, n0 = nt * 64;
  #pragma unroll
  for (int i = 0; i < 16; ++i) {
    int kk = rg * 16 + i;
    tile[kk * 66 + c] = (__bf16)W[(size_t)(k0 + kk) * HID + n0 + c];
  }
  __syncthreads();
  #pragma unroll
  for (int i = 0; i < 16; ++i) {
    int nn = rg * 16 + i;
    wt[(size_t)(g * HID + n0 + nn) * 2048 + k0 + c] = tile[c * 66 + nn];
  }
}

// --------------------------------------------------------------------- init
__global__ __launch_bounds__(256) void init_state(__bf16* __restrict__ hb0,
                                                  unsigned* __restrict__ bar) {
  int tid = blockIdx.x * 256 + threadIdx.x;
  if (tid == 0) *bar = 0u;
  for (int i = tid; i < BATCH * HID; i += gridDim.x * 256) hb0[i] = (__bf16)0.0f;
}

// ---------------------------------------------------------- persistent LSTM
__global__ __launch_bounds__(256, 1) void lstm_persistent(
    const __bf16* __restrict__ xb, const __bf16* __restrict__ wt,
    const float* __restrict__ bfp, const float* __restrict__ bip,
    const float* __restrict__ bcp, const float* __restrict__ bop,
    __bf16* __restrict__ hb0, __bf16* __restrict__ hb1,
    float* __restrict__ hf, unsigned* __restrict__ bar) {
  // two slabs (h rows, x rows), padded row stride 1032 elements
  __shared__ __bf16 lds[2 * SLAB_ELTS];

  const int m      = blockIdx.x & 3;
  const int jgroup = blockIdx.x >> 2;
  const int wave   = threadIdx.x >> 5;
  const int lane   = threadIdx.x & 31;
  const int r15    = lane & 15;
  const int hi     = lane >> 4;
  const int jbase  = (jgroup * 8 + wave) * 16;
  const int mbase  = m * 16;
  const int akbase = hi * 16;   // A: lanes 0-15 -> K 0-15; 16-31 -> K 16-31
  const int bkbase = hi * 16;   // B: lanes 0-15 -> K 0-15; 16-31 -> K 16-31

  const __bf16* wtg0 = wt + ((size_t)(0 * HID + jbase + r15) * 2048 + bkbase);
  const __bf16* wtg1 = wt + ((size_t)(1 * HID + jbase + r15) * 2048 + bkbase);
  const __bf16* wtg2 = wt + ((size_t)(2 * HID + jbase + r15) * 2048 + bkbase);
  const __bf16* wtg3 = wt + ((size_t)(3 * HID + jbase + r15) * 2048 + bkbase);

  const float bv0 = bfp[jbase + r15];
  const float bv1 = bip[jbase + r15];
  const float bv2 = bcp[jbase + r15];
  const float bv3 = bop[jbase + r15];

#if __has_builtin(__builtin_amdgcn_tensor_load_to_lds)
  // Constant part of the Tensor DMA descriptor (D# group 1):
  //   data_size=1 (2B), pad_enable=1, pad_interval=7 (256 DW), pad_amount=1 (2 DW)
  //   tensor_dim0 = tile_dim0 = 1024, tensor_dim1 = tile_dim1 = 16,
  //   tensor_dim0_stride = 1024.  Padding -> LDS row stride 1032 elements.
  v8i g1;
  g1[0] = (int)((1u << 16) | (1u << 20) | (7u << 22) | (1u << 25));
  g1[1] = (int)(1024u << 16);   // tensor_dim0[15:0]
  g1[2] = (int)(16u << 16);     // tensor_dim1[15:0]
  g1[3] = (int)(1024u << 16);   // tile_dim0
  g1[4] = (int)16;              // tile_dim1
  g1[5] = (int)1024;            // tensor_dim0_stride[31:0]
  g1[6] = 0;
  g1[7] = 0;
  const v4i gz4 = {0, 0, 0, 0};
  const v8i gz8 = {0, 0, 0, 0, 0, 0, 0, 0};
  const unsigned ldsBase = (unsigned)(uintptr_t)(&lds[0]);
#endif

  v8f cst;                                   // cell state lives in VGPRs
  #pragma unroll
  for (int e = 0; e < 8; ++e) cst[e] = 0.0f;

  for (int t = 0; t < SEQN; ++t) {
    const __bf16* hread  = (t & 1) ? hb1 : hb0;
    __bf16*       hwrite = (t & 1) ? hb0 : hb1;

    __syncthreads();                         // prior step's LDS reads done

#if __has_builtin(__builtin_amdgcn_tensor_load_to_lds)
    if (wave < 2) {                          // wave0: h slab, wave1: x_t slab
      const __bf16* src = (wave == 0)
          ? (hread + (size_t)mbase * HID)
          : (xb + ((size_t)t * BATCH + mbase) * HID);
      const unsigned long long ga = (unsigned long long)(uintptr_t)src;
      v4u g0;
      g0[0] = 1u;                                            // count=1
      g0[1] = ldsBase + (unsigned)(wave * SLAB_ELTS * 2);    // lds_addr (bytes)
      g0[2] = (unsigned)(ga & 0xFFFFFFFFull);                // global_addr[31:0]
      g0[3] = (unsigned)((ga >> 32) & 0x1FFFFFFull) | 0x80000000u; // [56:32]+type=2
      __builtin_amdgcn_tensor_load_to_lds(g0, g1, gz4, gz4, gz8, 0);
      __builtin_amdgcn_s_wait_tensorcnt(0);
    }
#else
    for (int c = threadIdx.x; c < 4096; c += 256) {   // 32 rows x 128 x 16B
      int row  = c >> 7;
      int coff = (c & 127) * 8;
      const __bf16* src = (row < 16)
          ? (hread + (size_t)(mbase + row) * HID + coff)
          : (xb + ((size_t)t * BATCH + (mbase + row - 16)) * HID + coff);
      *(uint4*)(&lds[row * LDS_ROW + coff + ((coff >= 512) ? 4 : 0)]) =
          *(const uint4*)src;
    }
#endif
    __syncthreads();

    v8f a0, a1, a2, a3;                      // f, i, c_hat, o accumulators
    #pragma unroll
    for (int e = 0; e < 8; ++e) { a0[e] = bv0; a1[e] = bv1; a2[e] = bv2; a3[e] = bv3; }

    for (int kt = 0; kt < 64; ++kt) {        // K = 2048 in steps of 32
      const int part = kt >> 5;              // 0: h-half, 1: x-half
      const int kk   = (kt & 31) * 32;
      const int pad4 = (kt & 16) ? 4 : 0;    // TDM pad: +4 elems past 512 in a row
      const __bf16* ap = &lds[part * SLAB_ELTS + r15 * LDS_ROW + kk + pad4 + akbase];
      v16bf af;
      ((uint4*)&af)[0] = *(const uint4*)ap;
      ((uint4*)&af)[1] = *(const uint4*)(ap + 8);
      const int koff = kt * 32;
      v16bf b0f, b1f, b2f, b3f;
      ((uint4*)&b0f)[0] = *(const uint4*)(wtg0 + koff);
      ((uint4*)&b0f)[1] = *(const uint4*)(wtg0 + koff + 8);
      ((uint4*)&b1f)[0] = *(const uint4*)(wtg1 + koff);
      ((uint4*)&b1f)[1] = *(const uint4*)(wtg1 + koff + 8);
      ((uint4*)&b2f)[0] = *(const uint4*)(wtg2 + koff);
      ((uint4*)&b2f)[1] = *(const uint4*)(wtg2 + koff + 8);
      ((uint4*)&b3f)[0] = *(const uint4*)(wtg3 + koff);
      ((uint4*)&b3f)[1] = *(const uint4*)(wtg3 + koff + 8);
      a0 = __builtin_amdgcn_wmma_f32_16x16x32_bf16(false, af, false, b0f, (short)0, a0, false, false);
      a1 = __builtin_amdgcn_wmma_f32_16x16x32_bf16(false, af, false, b1f, (short)0, a1, false, false);
      a2 = __builtin_amdgcn_wmma_f32_16x16x32_bf16(false, af, false, b2f, (short)0, a2, false, false);
      a3 = __builtin_amdgcn_wmma_f32_16x16x32_bf16(false, af, false, b3f, (short)0, a3, false, false);
    }

    #pragma unroll
    for (int e = 0; e < 8; ++e) {            // gate nonlinearities + state update
      float fg = sigf(a0[e]);
      float ig = sigf(a1[e]);
      float cg = tanh_fast(a2[e]);
      float og = sigf(a3[e]);
      float cn = fg * cst[e] + ig * cg;
      cst[e] = cn;
      float hn = og * tanh_fast(cn);
      int rowl = e + hi * 8;                 // C/D layout: vgpr e -> M=e (+8 hi half)
      size_t gidx = (size_t)(mbase + rowl) * HID + (jbase + r15);
      hwrite[gidx] = (__bf16)hn;
      if (t == SEQN - 1) hf[gidx] = hn;
    }

    // device-wide barrier (monotonic counter)
    __builtin_amdgcn_fence(__ATOMIC_RELEASE, "agent");
    __syncthreads();
    if (threadIdx.x == 0) {
      __hip_atomic_fetch_add(bar, 1u, __ATOMIC_RELAXED, __HIP_MEMORY_SCOPE_AGENT);
      const unsigned tgt = (unsigned)(t + 1) * gridDim.x;
      while (__hip_atomic_load(bar, __ATOMIC_RELAXED, __HIP_MEMORY_SCOPE_AGENT) < tgt) {
        __builtin_amdgcn_s_sleep(2);
      }
    }
    __syncthreads();
    __builtin_amdgcn_fence(__ATOMIC_ACQUIRE, "agent");
  }
}

// --------------------------------------------------------------- classifier
__global__ __launch_bounds__(128) void fc_head(const float* __restrict__ hfin,
                                               const float* __restrict__ fcW,
                                               const float* __restrict__ fcb,
                                               float* __restrict__ out) {
  int tid = threadIdx.x;                     // 128 = 64 batch x 2 classes
  int b = tid >> 1, c = tid & 1;
  float s = fcb[c];
  for (int k = 0; k < HID; ++k) s += hfin[(size_t)b * HID + k] * fcW[k * 2 + c];
  out[b * 2 + c] = s;
}

// ------------------------------------------------------------------- launch
extern "C" void kernel_launch(void* const* d_in, const int* in_sizes, int n_in,
                              void* d_out, int out_size, void* d_ws, size_t ws_size,
                              hipStream_t stream) {
  (void)in_sizes; (void)n_in; (void)out_size; (void)ws_size;
  const int*   xi  = (const int*)d_in[0];
  const float* emb = (const float*)d_in[1];
  const float* Wf  = (const float*)d_in[2];
  const float* bf_ = (const float*)d_in[3];
  const float* Wi  = (const float*)d_in[4];
  const float* bi_ = (const float*)d_in[5];
  const float* Wc  = (const float*)d_in[6];
  const float* bc_ = (const float*)d_in[7];
  const float* Wo  = (const float*)d_in[8];
  const float* bo_ = (const float*)d_in[9];
  const float* fcW = (const float*)d_in[10];
  const float* fcb = (const float*)d_in[11];
  float* out = (float*)d_out;

  char* ws = (char*)d_ws;
  __bf16*   xb  = (__bf16*)(ws + X_OFF);
  __bf16*   wt  = (__bf16*)(ws + WT_OFF);
  __bf16*   hb0 = (__bf16*)(ws + H0_OFF);
  __bf16*   hb1 = (__bf16*)(ws + H1_OFF);
  float*    hf  = (float*)(ws + HF_OFF);
  unsigned* bar = (unsigned*)(ws + BAR_OFF);

  embed_cast<<<dim3(32768), dim3(256), 0, stream>>>(xi, emb, xb);
  pack_weights<<<dim3(2048), dim3(256), 0, stream>>>(Wf, Wi, Wc, Wo, wt);
  init_state<<<dim3(32), dim3(256), 0, stream>>>(hb0, bar);
  lstm_persistent<<<dim3(32), dim3(256), 0, stream>>>(xb, wt, bf_, bi_, bc_, bo_,
                                                      hb0, hb1, hf, bar);
  fc_head<<<dim3(1), dim3(128), 0, stream>>>(hf, fcW, fcb, out);
}
